// TabNet_57878979281018
// MI455X (gfx1250) — matrix operand order repaired and checked by reading, not compiled, not run on hardware
//
#include <hip/hip_runtime.h>
#include <hip/hip_bf16.h>
#include <stdint.h>

// ---------------------------------------------------------------------------
// TabNet forward for MI455X (gfx1250, wave32, WMMA).
// GEMMs: v_wmma_f32_16x16x32_f16, A/W staged through LDS (fp32 -> f16).
// Template KP/NP so fragment arrays stay in VGPRs (no scratch spills) and
// fragments load as ds_load_b128 pairs. BN stats via deterministic two-stage
// column reduction. Sparsemax computed sort-free per row (one wave32/row).
// ---------------------------------------------------------------------------

#define B_ROWS 65536
#define N_NUM  54
#define N_CAT  10
#define EMB_D  5
#define D_IN   104      // 54 + 10*5
#define ND     64
#define NA     64
#define H_DIM  128      // ND + NA
#define H2_DIM 256      // 2*(ND+NA)
#define NSTEPS 6
#define RELAX_C 1.5f
#define EPS_C   1e-5f
#define SCALE_C 0.7071067811865476f

typedef __attribute__((ext_vector_type(16))) _Float16 v16h;
typedef __attribute__((ext_vector_type(8)))  float    v8f;

union Frag { uint32_t u[8]; v16h h; };
union PK   { _Float16 h[2]; uint32_t u; };

// ---------------------------------------------------------------------------
// C(B x N) = A(B x K, row stride lda, fp32) * W(K x N, fp32)
// KP = K padded to mult of 32 (compile-time); NP = N padded to mult of 16.
// 256 threads = 8 wave32s; each block computes 128 rows x NP cols.
//
// sA: row-major f16, skewed stride KP+8 halfs (row stride 272/144 B, 16B
//     aligned) -> A fragment = two ds_load_b128 per K-slice.
// sW: fragment-major swizzle: sW[((nt*NKT+kt)*32 + lane)*8 + j] holds the
//     packed pair (W[k][n], W[k+1][n]) with k = kt*32 + 2j + 16*(lane>>4),
//     n = nt*16 + (lane&15)  -> B fragment = two ds_load_b128, no VALU math.
// ---------------------------------------------------------------------------
template<int KP, int NP>
__global__ __launch_bounds__(256) void gemm_f16_wmma(
    const float* __restrict__ A, int lda,
    const float* __restrict__ W,
    float* __restrict__ C,
    int K, int N)
{
  constexpr int NKT = KP / 32;          // K-slices of 32
  constexpr int NT  = NP / 16;          // 16-col tiles
  constexpr int SA_STRIDE = KP + 8;     // halfs

  __shared__ __align__(16) _Float16 sA[128 * SA_STRIDE];
  __shared__ __align__(16) uint32_t sW[NT * NKT * 256];

  const int tid = threadIdx.x;
  const int r0  = blockIdx.x * 128;

  // ---- Stage A panel (fp32 -> f16), zero-pad K..KP ----
  for (int idx = tid; idx < 128 * KP; idx += 256) {
    int r = idx / KP, k = idx - r * KP;
    float v = (k < K) ? A[(size_t)(r0 + r) * lda + k] : 0.f;
    sA[r * SA_STRIDE + k] = (_Float16)v;
  }

  // ---- Stage W: coalesced global reads, swizzled LDS writes ----
  for (int idx = tid; idx < (KP / 2) * NP; idx += 256) {
    int kp = idx / NP, n = idx - kp * NP;     // pair index, column
    int k  = kp * 2;
    PK p;
    p.h[0] = (_Float16)((k     < K && n < N) ? W[(size_t)k       * N + n] : 0.f);
    p.h[1] = (_Float16)((k + 1 < K && n < N) ? W[(size_t)(k + 1) * N + n] : 0.f);
    int kt   = k >> 5;
    int ko   = k - kt * 32;                   // 0..30, even
    int hh   = ko >> 4;                       // 16*half component
    int j    = (ko & 15) >> 1;
    int nt   = n >> 4;
    int lane = (hh << 4) | (n & 15);
    sW[(((nt * NKT + kt) * 32) + lane) * 8 + j] = p.u;
  }
  __syncthreads();

  const int wave  = tid >> 5;
  const int lane  = tid & 31;
  const int half_ = lane >> 4;
  const int mrow  = lane & 15;
  const int m0    = wave * 16;

  // ---- Preload A fragments (registers; NKT is constexpr) ----
  Frag afr[NKT];
  const uint4* pA = (const uint4*)(sA + (m0 + mrow) * SA_STRIDE);
  #pragma unroll
  for (int kt = 0; kt < NKT; ++kt) {
    uint4 lo = pA[kt * 4 + 2 * half_];    // u32 idx kt*16 + 8*half
    uint4 hi = pA[kt * 4 + 2 * half_ + 1];// u32 idx kt*16 + 8*half + 4
    afr[kt].u[0] = lo.x; afr[kt].u[1] = lo.y; afr[kt].u[2] = lo.z; afr[kt].u[3] = lo.w;
    afr[kt].u[4] = hi.x; afr[kt].u[5] = hi.y; afr[kt].u[6] = hi.z; afr[kt].u[7] = hi.w;
  }

  const uint4* pW = (const uint4*)sW;
  for (int nt = 0; nt < NT; ++nt) {
    const int n = nt * 16 + mrow;
    v8f acc = {};
    #pragma unroll
    for (int kt = 0; kt < NKT; ++kt) {
      const int fbase = ((nt * NKT + kt) * 32 + lane) * 2;   // uint4 index
      uint4 lo = pW[fbase];
      uint4 hi = pW[fbase + 1];
      Frag bf;
      bf.u[0] = lo.x; bf.u[1] = lo.y; bf.u[2] = lo.z; bf.u[3] = lo.w;
      bf.u[4] = hi.x; bf.u[5] = hi.y; bf.u[6] = hi.z; bf.u[7] = hi.w;
      acc = __builtin_amdgcn_wmma_f32_16x16x32_f16(
          false, afr[kt].h, false, bf.h, (short)0, acc, false, false);
    }
    if (n < N) {
      const int rowbase = r0 + m0 + 8 * half_;
      #pragma unroll
      for (int r = 0; r < 8; ++r)
        C[(size_t)(rowbase + r) * N + n] = acc[r];
    }
  }
}

// ---------------------------------------------------------------------------
// Deterministic column stats: stage 1 partial sums, stage 2 finalize.
// ---------------------------------------------------------------------------
__global__ void colstats_partial(const float* __restrict__ Y, int N,
                                 int rowsPerBlock, float* __restrict__ part)
{
  int c = threadIdx.x;
  if (c >= N) return;
  int r0 = blockIdx.x * rowsPerBlock;
  float s = 0.f, q = 0.f;
  for (int r = 0; r < rowsPerBlock; ++r) {
    float v = Y[(size_t)(r0 + r) * N + c];
    s += v; q += v * v;
  }
  part[(size_t)blockIdx.x * 2 * N + c]     = s;
  part[(size_t)blockIdx.x * 2 * N + N + c] = q;
}

__global__ void colstats_final(const float* __restrict__ part, int nblocks,
                               int N, float invB, float* __restrict__ stats)
{
  int c = threadIdx.x;
  if (c >= N) return;
  float s = 0.f, q = 0.f;
  for (int b = 0; b < nblocks; ++b) {
    s += part[(size_t)b * 2 * N + c];
    q += part[(size_t)b * 2 * N + N + c];
  }
  float m = s * invB;
  float v = q * invB - m * m;
  stats[c]     = m;
  stats[N + c] = rsqrtf(v + EPS_C);
}

// ---------------------------------------------------------------------------
// GLU epilogue: H = (alpha*Hprev + BN(a)*sigmoid(BN(c))) * scale
// Y is B x 256 (a = cols 0..127, c = cols 128..255).
// ---------------------------------------------------------------------------
__global__ void act_glu(const float* __restrict__ Y, const float* __restrict__ stats,
                        const float* __restrict__ g, const float* __restrict__ be,
                        const float* __restrict__ Hprev, float alpha, float scale,
                        float* __restrict__ Hout)
{
  int gid = blockIdx.x * blockDim.x + threadIdx.x;
  int r = gid >> 7, c = gid & 127;
  size_t yb = (size_t)r * H2_DIM;
  float a  = (Y[yb + c]  - stats[c])  * stats[H2_DIM + c]  * g[c]  + be[c];
  int cc = c + 128;
  float cv = (Y[yb + cc] - stats[cc]) * stats[H2_DIM + cc] * g[cc] + be[cc];
  float h  = a * (1.f / (1.f + __expf(-cv)));
  float prev = (alpha != 0.f) ? Hprev[gid] : 0.f;
  Hout[gid] = (prev * alpha + h) * scale;
}

// ---------------------------------------------------------------------------
// Embedding lookup + concat into X0 (B x 104).
// ---------------------------------------------------------------------------
__global__ void embed_concat(const float* __restrict__ numeric,
                             const int* __restrict__ cats,
                             const float* __restrict__ emb,
                             float* __restrict__ X0)
{
  int gid = blockIdx.x * blockDim.x + threadIdx.x;
  int r = gid / D_IN, c = gid - r * D_IN;
  float v;
  if (c < N_NUM) {
    v = numeric[(size_t)r * N_NUM + c];
  } else {
    int j = (c - N_NUM) / EMB_D;
    int e = (c - N_NUM) - j * EMB_D;
    int idx = cats[(size_t)r * N_CAT + j];
    v = emb[((size_t)j * 100 + idx) * EMB_D + e];
  }
  X0[gid] = v;
}

__global__ void bn_apply(float* __restrict__ X, const float* __restrict__ stats,
                         const float* __restrict__ g, const float* __restrict__ b, int N)
{
  int gid = blockIdx.x * blockDim.x + threadIdx.x;
  int c = gid % N;
  X[gid] = (X[gid] - stats[c]) * stats[N + c] * g[c] + b[c];
}

// ---------------------------------------------------------------------------
// Per-row sparsemax (sort-free, exact) + prior update + masked input.
// One wave32 per row; 8 rows per block.
// z = BN(Y)*prior; k = #{i: f(z_i)<1}, f(v)=sum(max(z_j-v,0));
// tau = (sum_{f<1} z - 1)/k; mask = max(z - tau, 0).
// ---------------------------------------------------------------------------
__global__ __launch_bounds__(256) void sparsemax_prior(
    const float* __restrict__ Y, const float* __restrict__ stats,
    const float* __restrict__ g, const float* __restrict__ be,
    float* __restrict__ prior, const float* __restrict__ origin,
    float* __restrict__ maskedX)
{
  __shared__ float zl[8][112];
  int wave = threadIdx.x >> 5, lane = threadIdx.x & 31;
  int row  = blockIdx.x * 8 + wave;
  size_t base = (size_t)row * D_IN;

  float zown[4], pown[4];
  #pragma unroll
  for (int i = 0; i < 4; ++i) {
    int j = lane + 32 * i;
    if (j < D_IN) {
      float bn = (Y[base + j] - stats[j]) * stats[D_IN + j] * g[j] + be[j];
      float p  = prior[base + j];
      float z  = bn * p;
      zown[i] = z; pown[i] = p;
      zl[wave][j] = z;
    } else { zown[i] = -1e30f; pown[i] = 0.f; }
  }
  __syncthreads();

  float f[4] = {0.f, 0.f, 0.f, 0.f};
  for (int j = 0; j < D_IN; ++j) {
    float zj = zl[wave][j];
    #pragma unroll
    for (int i = 0; i < 4; ++i) f[i] += fmaxf(zj - zown[i], 0.f);
  }
  float cnt = 0.f, s = 0.f;
  #pragma unroll
  for (int i = 0; i < 4; ++i)
    if (f[i] < 1.f) { cnt += 1.f; s += zown[i]; }
  #pragma unroll
  for (int m = 16; m >= 1; m >>= 1) {
    cnt += __shfl_xor(cnt, m, 32);
    s   += __shfl_xor(s,   m, 32);
  }
  float tau = (s - 1.f) / cnt;

  #pragma unroll
  for (int i = 0; i < 4; ++i) {
    int j = lane + 32 * i;
    if (j < D_IN) {
      float msk = fmaxf(zown[i] - tau, 0.f);
      prior[base + j]   = pown[i] * (RELAX_C - msk);
      maskedX[base + j] = msk * origin[base + j];
    }
  }
}

__global__ void final_accum(const float* __restrict__ H, float* __restrict__ F)
{
  int gid = blockIdx.x * blockDim.x + threadIdx.x;
  int r = gid >> 6, c = gid & 63;
  F[gid] += fmaxf(H[(size_t)r * H_DIM + c], 0.f);
}

__global__ void out_kernel(const float* __restrict__ F, const float* __restrict__ fw,
                           const float* __restrict__ fb, float* __restrict__ out)
{
  int r = blockIdx.x * blockDim.x + threadIdx.x;
  float s = 0.f;
  #pragma unroll 8
  for (int i = 0; i < ND; ++i) s += F[(size_t)r * ND + i] * fw[i];
  out[r] = s + fb[0];
}

__global__ void fill_kernel(float* __restrict__ p, float v, int n)
{
  int gid = blockIdx.x * blockDim.x + threadIdx.x;
  if (gid < n) p[gid] = v;
}

// ---------------------------------------------------------------------------
// Host orchestration (all on `stream`, graph-capture safe).
// Input layout assumption: depth-first insertion-order flattening of
// setup_inputs(): numeric, cats, then params leaves (95 arrays total).
// ---------------------------------------------------------------------------
extern "C" void kernel_launch(void* const* d_in, const int* in_sizes, int n_in,
                              void* d_out, int out_size, void* d_ws, size_t ws_size,
                              hipStream_t stream)
{
  (void)in_sizes; (void)n_in; (void)out_size; (void)ws_size;

  const float* numeric = (const float*)d_in[0];
  const int*   cats    = (const int*)  d_in[1];
  const float* emb     = (const float*)d_in[2];
  const float* fbn_g   = (const float*)d_in[3];
  const float* fbn_b   = (const float*)d_in[4];
  // shared[0]: W,b,g,be = 5..8 ; shared[1]: 9..12 ; ft[0]: 13..16 ; ft[1]: 17..20
  const float* shW[2]  = {(const float*)d_in[5],  (const float*)d_in[9]};
  const float* shG[2]  = {(const float*)d_in[7],  (const float*)d_in[11]};
  const float* shBe[2] = {(const float*)d_in[8],  (const float*)d_in[12]};
  const float* ftW[2]  = {(const float*)d_in[13], (const float*)d_in[17]};
  const float* ftG[2]  = {(const float*)d_in[15], (const float*)d_in[19]};
  const float* ftBe[2] = {(const float*)d_in[16], (const float*)d_in[20]};
  const float* finW = (const float*)d_in[21 + NSTEPS * 12];
  const float* finB = (const float*)d_in[22 + NSTEPS * 12];

  // Workspace partition (floats)
  float* p = (float*)d_ws;
  float* origin  = p; p += (size_t)B_ROWS * D_IN;
  float* maskedX = p; p += (size_t)B_ROWS * D_IN;
  float* prior   = p; p += (size_t)B_ROWS * D_IN;
  float* Y       = p; p += (size_t)B_ROWS * H2_DIM;
  float* H       = p; p += (size_t)B_ROWS * H_DIM;
  float* Fin     = p; p += (size_t)B_ROWS * ND;
  float* part    = p; p += (size_t)256 * 2 * H2_DIM;
  float* stats   = p; p += 2 * H2_DIM;

  const int ROWS_PER_STAT_BLOCK = B_ROWS / 256;  // 256 partial blocks

  auto stats_for = [&](const float* Ybuf, int N) {
    colstats_partial<<<256, 256, 0, stream>>>(Ybuf, N, ROWS_PER_STAT_BLOCK, part);
    colstats_final<<<1, 256, 0, stream>>>(part, 256, N, 1.f / (float)B_ROWS, stats);
  };
  auto glu = [&](const float* Ap, int lda, int K,
                 const float* Wm, const float* g, const float* be,
                 float alpha, float scale) {
    gemm_f16_wmma<128, 256><<<B_ROWS / 128, 256, 0, stream>>>(Ap, lda, Wm, Y, K, H2_DIM);
    stats_for(Y, H2_DIM);
    act_glu<<<(B_ROWS * H_DIM) / 256, 256, 0, stream>>>(Y, stats, g, be, H, alpha, scale, H);
  };

  // --- feature BN over concat(numeric, emb) ---
  embed_concat<<<(B_ROWS * D_IN) / 256, 256, 0, stream>>>(numeric, cats, emb, origin);
  stats_for(origin, D_IN);
  bn_apply<<<(B_ROWS * D_IN) / 256, 256, 0, stream>>>(origin, stats, fbn_g, fbn_b, D_IN);
  fill_kernel<<<(B_ROWS * D_IN) / 256, 256, 0, stream>>>(prior, 1.f, B_ROWS * D_IN);
  fill_kernel<<<(B_ROWS * ND)   / 256, 256, 0, stream>>>(Fin,   0.f, B_ROWS * ND);

  // --- initial trunk: shared_ft then step_ft ---
  glu(origin, D_IN, D_IN, shW[0], shG[0], shBe[0], 0.f, 1.f);
  glu(H, H_DIM, H_DIM, shW[1], shG[1], shBe[1], 1.f, SCALE_C);
  glu(H, H_DIM, H_DIM, ftW[0], ftG[0], ftBe[0], 1.f, SCALE_C);
  glu(H, H_DIM, H_DIM, ftW[1], ftG[1], ftBe[1], 1.f, SCALE_C);

  // --- decision steps ---
  for (int s = 0; s < NSTEPS; ++s) {
    int base = 21 + s * 12;
    const float* attW  = (const float*)d_in[base + 0];
    const float* attG  = (const float*)d_in[base + 2];
    const float* attBe = (const float*)d_in[base + 3];
    const float* f0W   = (const float*)d_in[base + 4];
    const float* f0G   = (const float*)d_in[base + 6];
    const float* f0Be  = (const float*)d_in[base + 7];
    const float* f1W   = (const float*)d_in[base + 8];
    const float* f1G   = (const float*)d_in[base + 10];
    const float* f1Be  = (const float*)d_in[base + 11];

    // attention projection: att (= H[:,64:]) @ attW(64x104) -> Y (ld 104)
    gemm_f16_wmma<64, 112><<<B_ROWS / 128, 256, 0, stream>>>(H + NA, H_DIM, attW, Y,
                                                             NA, D_IN);
    stats_for(Y, D_IN);
    sparsemax_prior<<<B_ROWS / 8, 256, 0, stream>>>(Y, stats, attG, attBe,
                                                    prior, origin, maskedX);

    glu(maskedX, D_IN, D_IN, shW[0], shG[0], shBe[0], 0.f, 1.f);
    glu(H, H_DIM, H_DIM, shW[1], shG[1], shBe[1], 1.f, SCALE_C);
    glu(H, H_DIM, H_DIM, f0W, f0G, f0Be, 1.f, SCALE_C);
    glu(H, H_DIM, H_DIM, f1W, f1G, f1Be, 1.f, SCALE_C);

    final_accum<<<(B_ROWS * ND) / 256, 256, 0, stream>>>(H, Fin);
  }

  out_kernel<<<B_ROWS / 256, 256, 0, stream>>>(Fin, finW, finB, (float*)d_out);
}
